// CrossDeformableAttention_29205777613323
// MI455X (gfx1250) — matrix operand order repaired and checked by reading, not compiled, not run on hardware
//
#include <hip/hip_runtime.h>
#include <hip/hip_bf16.h>

// ---------------------------------------------------------------------------
// CrossDeformableAttention for MI455X (gfx1250).
// wave32, v_wmma_f32_16x16x32_bf16 GEMMs, global_load_async_to_lds staging.
// Shapes: B=4, C=256, HEADS=8, POINTS=8, W=H=64, T=nq=4096, hd=32, K=256.
// ---------------------------------------------------------------------------

#define TTOK 4096
#define KDIM 256
#define NBATCH 4

typedef __attribute__((ext_vector_type(16))) __bf16        v16bf;
typedef __attribute__((ext_vector_type(8)))  float         v8f;
typedef __attribute__((ext_vector_type(8)))  unsigned int  v8u;

__device__ __forceinline__ unsigned short f2bf(float f) {
    unsigned int u = __builtin_bit_cast(unsigned int, f);
    unsigned int r = (u + 0x7FFFu + ((u >> 16) & 1u)) >> 16;  // RNE f32->bf16
    return (unsigned short)r;
}

// Async global->LDS copy of 16 bytes per lane (ASYNCcnt-tracked TDM-lite path).
__device__ __forceinline__ void async_ldg_b128(unsigned lds_byte_off,
                                               const float* gaddr) {
    asm volatile("global_load_async_to_lds_b128 %0, %1, off"
                 :: "v"(lds_byte_off), "v"(gaddr)
                 : "memory");
}

__device__ __forceinline__ void wait_async0() {
#if __has_builtin(__builtin_amdgcn_s_wait_asynccnt)
    __builtin_amdgcn_s_wait_asynccnt(0);
#else
    asm volatile("s_wait_asynccnt 0" ::: "memory");
#endif
}

// LDS byte offset of a __shared__ object: low 32 bits of its flat address.
template <typename T>
__device__ __forceinline__ unsigned lds_off(const T* p) {
    return (unsigned)(size_t)p;
}

// ---------------------------------------------------------------------------
// GEMM: out(b,t,n) = A(b,t,:) @ W(:,n) + bias(n)  [+ residual, various layouts]
//   amode 0: A channel-major  A[(b*K + k)*T + t]   (query/value raw layout)
//   amode 1: A row-major      A[(b*T + t)*K + k]   (workspace layout)
//   emode 0: out[(b*T+t)*N + n] = acc + bias
//   emode 1: out[(b*T+t)*N + n] = acc + bias + resid[(b*C+n)*T + t]
//   emode 2: out[(b*C+n)*T + t] = acc + bias + resid[(b*C+n)*T + t]
// Block: 256 threads = 8 waves. Block tile: 128 tokens x 64 cols, K chunk 32.
// Wave w owns token rows [w*16, w*16+16) and all 64 cols: 4 WMMAs per chunk
// sharing one A fragment.
// ---------------------------------------------------------------------------
__global__ __launch_bounds__(256) void msda_gemm(
    const float* __restrict__ A, const float* __restrict__ W,
    const float* __restrict__ bias, const float* __restrict__ resid,
    float* __restrict__ out, int N, int amode, int emode)
{
    __shared__ alignas(16) float          Af[128 * 32];       // f32 stage (16KB)
    __shared__ alignas(16) float          Bf[32 * 64];        // f32 stage (8KB)
    __shared__ alignas(16) unsigned short As[128 * 32];       // bf16 [t][k] (8KB)
    __shared__ alignas(16) unsigned short Bs[64 * 32];        // bf16 [n][k] (4KB)

    const int tid  = threadIdx.x;
    const int lane = tid & 31;
    const int w    = tid >> 5;        // wave id 0..7 -> token strip
    const int hi   = lane >> 4;
    const int nl   = lane & 15;
    const int m    = lane & 15;
    const int t0   = blockIdx.x * 128;
    const int n0   = blockIdx.y * 64;
    const int b    = blockIdx.z;

    const float* Ab = A + (size_t)b * KDIM * TTOK;   // K*T == T*K per batch
    const unsigned int* As32 = (const unsigned int*)As;
    const unsigned int* Bs32 = (const unsigned int*)Bs;

    v8f acc0 = {}, acc1 = {}, acc2 = {}, acc3 = {};

    #pragma unroll
    for (int k0 = 0; k0 < KDIM; k0 += 32) {
        __syncthreads();  // previous chunk's fragments fully consumed

        // ---- async stage A tile (f32): 1024 x 16B units ----
        if (amode == 0) {
            // Af[k][t] : 32 rows x 128 floats
            #pragma unroll
            for (int r = 0; r < 4; ++r) {
                const int unit = tid + r * 256;
                const int row = unit >> 5, col = unit & 31;
                async_ldg_b128(lds_off(Af) + unit * 16,
                               Ab + (size_t)(k0 + row) * TTOK + t0 + col * 4);
            }
        } else {
            // Af[t][k] : 128 rows x 32 floats
            #pragma unroll
            for (int r = 0; r < 4; ++r) {
                const int unit = tid + r * 256;
                const int row = unit >> 3, col = unit & 7;
                async_ldg_b128(lds_off(Af) + unit * 16,
                               Ab + (size_t)(t0 + row) * KDIM + k0 + col * 4);
            }
        }
        // ---- async stage W tile (f32): Bf[k][n] 32 x 64, 512 units ----
        #pragma unroll
        for (int r = 0; r < 2; ++r) {
            const int unit = tid + r * 256;
            const int row = unit >> 4, col = unit & 15;
            async_ldg_b128(lds_off(Bf) + unit * 16,
                           W + (size_t)(k0 + row) * N + n0 + col * 4);
        }
        wait_async0();
        __syncthreads();  // Af/Bf visible to all waves

        // ---- convert f32 -> bf16 into WMMA-packed layouts ----
        if (amode == 0) {
            // transpose Af[k][t] -> As[t][k]
            const int tl = tid >> 1;
            const int kq = (tid & 1) * 16;
            #pragma unroll
            for (int i = 0; i < 16; ++i)
                As[tl * 32 + kq + i] = f2bf(Af[(kq + i) * 128 + tl]);
        } else {
            // straight convert, both [t][k]
            const int idx = tid * 16;
            #pragma unroll
            for (int i = 0; i < 16; ++i)
                As[idx + i] = f2bf(Af[idx + i]);
        }
        {
            // transpose Bf[k][n] -> Bs[n][k]
            const int nr = tid >> 2;
            const int kq = (tid & 3) * 8;
            #pragma unroll
            for (int i = 0; i < 8; ++i)
                Bs[nr * 32 + kq + i] = f2bf(Bf[(kq + i) * 64 + nr]);
        }
        __syncthreads();  // As/Bs ready

        // ---- fragments: one aligned 32-bit LDS read per packed K pair ----
        v8u au, bu0, bu1, bu2, bu3;
        #pragma unroll
        for (int j = 0; j < 8; ++j) {
            // A K-pair dword: (j/4)*8 + hi*4 + (j%4)
            const int ka2 = ((j >> 2) << 3) + (hi << 2) + (j & 3);
            au[j] = As32[(w * 16 + m) * 16 + ka2];
            // B K-pair dword: (j/4)*8 + hi*4 + (j%4)
            const int kb2 = ((j >> 2) << 3) + (hi << 2) + (j & 3);
            bu0[j] = Bs32[(0 * 16 + nl) * 16 + kb2];
            bu1[j] = Bs32[(1 * 16 + nl) * 16 + kb2];
            bu2[j] = Bs32[(2 * 16 + nl) * 16 + kb2];
            bu3[j] = Bs32[(3 * 16 + nl) * 16 + kb2];
        }
        const v16bf av = __builtin_bit_cast(v16bf, au);
        acc0 = __builtin_amdgcn_wmma_f32_16x16x32_bf16(
            false, av, false, __builtin_bit_cast(v16bf, bu0), (short)0, acc0, false, false);
        acc1 = __builtin_amdgcn_wmma_f32_16x16x32_bf16(
            false, av, false, __builtin_bit_cast(v16bf, bu1), (short)0, acc1, false, false);
        acc2 = __builtin_amdgcn_wmma_f32_16x16x32_bf16(
            false, av, false, __builtin_bit_cast(v16bf, bu2), (short)0, acc2, false, false);
        acc3 = __builtin_amdgcn_wmma_f32_16x16x32_bf16(
            false, av, false, __builtin_bit_cast(v16bf, bu3), (short)0, acc3, false, false);
    }

    // ---- epilogue: 4 subtiles of 16x16 per wave ----
    #pragma unroll
    for (int s = 0; s < 4; ++s) {
        const v8f acc = (s == 0) ? acc0 : (s == 1) ? acc1 : (s == 2) ? acc2 : acc3;
        const int n = n0 + s * 16 + nl;
        const float bvn = bias[n];
        #pragma unroll
        for (int r = 0; r < 8; ++r) {
            const int t = t0 + w * 16 + r + (hi << 3);
            float v = acc[r] + bvn;
            if (emode == 0) {
                out[((size_t)b * TTOK + t) * N + n] = v;
            } else if (emode == 1) {
                v += resid[((size_t)b * KDIM + n) * TTOK + t];
                out[((size_t)b * TTOK + t) * N + n] = v;
            } else {
                v += resid[((size_t)b * KDIM + n) * TTOK + t];
                out[((size_t)b * KDIM + n) * TTOK + t] = v;
            }
        }
    }
}

// ---------------------------------------------------------------------------
// Sampling: one wave per (b, head, token); 32 lanes = the 32 head channels.
// Inline softmax over P=8, bilinear zero-pad gather from val(b,t,256).
// ---------------------------------------------------------------------------
__global__ __launch_bounds__(256) void msda_sample(
    const float* __restrict__ val,   // (b, t, 256)
    const float* __restrict__ off,   // (b, t, 128) = (H,P,2)
    const float* __restrict__ attn,  // (b, t, 64)  = (H,P) logits
    float* __restrict__ samp)        // (b, t, 256)
{
    const int lane = threadIdx.x & 31;
    const int wid  = threadIdx.x >> 5;
    const int gw   = blockIdx.x * 8 + wid;     // (b*8 + head)*4096 + t
    const int t    = gw & 4095;
    const int head = (gw >> 12) & 7;
    const int b    = gw >> 15;

    const int i = t >> 6;
    const int j = t & 63;
    const float refx = (float)i * (1.0f / 63.0f);
    const float refy = (float)j * (1.0f / 63.0f);

    const float* offp  = off  + ((size_t)b * TTOK + t) * 128 + head * 16;
    const float* attnp = attn + ((size_t)b * TTOK + t) * 64  + head * 8;
    const float* valb  = val  + (size_t)b * TTOK * 256 + head * 32 + lane;

    float lg[8];
    float mx = -1e30f;
    #pragma unroll
    for (int p = 0; p < 8; ++p) { lg[p] = attnp[p]; mx = fmaxf(mx, lg[p]); }
    float se = 0.0f;
    #pragma unroll
    for (int p = 0; p < 8; ++p) { lg[p] = __expf(lg[p] - mx); se += lg[p]; }
    const float inv = 1.0f / se;

    auto corner = [&](int yi, int xi) -> float {
        if ((unsigned)yi < 64u && (unsigned)xi < 64u)
            return valb[(size_t)(yi * 64 + xi) * 256];
        return 0.0f;
    };

    float acc = 0.0f;
    #pragma unroll
    for (int p = 0; p < 8; ++p) {
        const float ox = offp[p * 2 + 0];
        const float oy = offp[p * 2 + 1];
        const float x = (refx + ox * (1.0f / 64.0f)) * 64.0f - 0.5f;
        const float y = (refy + oy * (1.0f / 64.0f)) * 64.0f - 0.5f;
        const float x0f = floorf(x), y0f = floorf(y);
        const float wx = x - x0f, wy = y - y0f;
        const int x0 = (int)x0f, y0 = (int)y0f;
        const float a = lg[p] * inv;
        acc += a * (corner(y0,     x0    ) * (1.0f - wy) * (1.0f - wx)
                  + corner(y0,     x0 + 1) * (1.0f - wy) * wx
                  + corner(y0 + 1, x0    ) * wy * (1.0f - wx)
                  + corner(y0 + 1, x0 + 1) * wy * wx);
    }
    samp[((size_t)b * TTOK + t) * 256 + head * 32 + lane] = acc;
}

// ---------------------------------------------------------------------------
extern "C" void kernel_launch(void* const* d_in, const int* in_sizes, int n_in,
                              void* d_out, int out_size, void* d_ws, size_t ws_size,
                              hipStream_t stream) {
    const float* query = (const float*)d_in[0];
    const float* value = (const float*)d_in[1];
    const float* Wv    = (const float*)d_in[2];
    const float* bv    = (const float*)d_in[3];
    const float* Wo    = (const float*)d_in[4];
    const float* bo    = (const float*)d_in[5];
    const float* Wa    = (const float*)d_in[6];
    const float* ba    = (const float*)d_in[7];
    const float* Wi    = (const float*)d_in[8];
    const float* bi    = (const float*)d_in[9];
    const float* Wout  = (const float*)d_in[10];
    const float* bout  = (const float*)d_in[11];
    float* out = (float*)d_out;

    float* ws      = (float*)d_ws;
    float* valbuf  = ws;                                        // 4*4096*256
    float* offbuf  = valbuf  + (size_t)NBATCH * TTOK * 256;     // 4*4096*128
    float* attnbuf = offbuf  + (size_t)NBATCH * TTOK * 128;     // 4*4096*64
    float* sampbuf = attnbuf + (size_t)NBATCH * TTOK * 64;      // 4*4096*256
    float* out1buf = sampbuf + (size_t)NBATCH * TTOK * 256;     // 4*4096*256

    const dim3 blk(256);
    // val = value @ Wv + bv
    msda_gemm<<<dim3(TTOK / 128, 4, NBATCH), blk, 0, stream>>>(
        value, Wv, bv, nullptr, valbuf, 256, 0, 0);
    // off = query @ Wo + bo
    msda_gemm<<<dim3(TTOK / 128, 2, NBATCH), blk, 0, stream>>>(
        query, Wo, bo, nullptr, offbuf, 128, 0, 0);
    // attn logits = query @ Wa + ba
    msda_gemm<<<dim3(TTOK / 128, 1, NBATCH), blk, 0, stream>>>(
        query, Wa, ba, nullptr, attnbuf, 64, 0, 0);
    // softmax + bilinear sampling -> samp(b,t,c)
    msda_sample<<<(NBATCH * 8 * TTOK) / 8, 256, 0, stream>>>(
        valbuf, offbuf, attnbuf, sampbuf);
    // out1 = samp @ Wi + bi + q
    msda_gemm<<<dim3(TTOK / 128, 4, NBATCH), blk, 0, stream>>>(
        sampbuf, Wi, bi, query, out1buf, 256, 1, 1);
    // d_out = out1 @ Wout + bout + value   (channel-major output)
    msda_gemm<<<dim3(TTOK / 128, 4, NBATCH), blk, 0, stream>>>(
        out1buf, Wout, bout, value, out, 256, 1, 2);
}
